// MoESineLayer_14800457302365
// MI455X (gfx1250) — matrix operand order repaired and checked by reading, not compiled, not run on hardware
//
#include <hip/hip_runtime.h>

// ---------------------------------------------------------------------------
// MoE sine layer for MI455X (gfx1250, wave32, WMMA).
//
// out[n] = sum_{k in top2} w_k * sin(30 * (x@W[e]+b[e]) * scale + shift)
//   [scale|shift] = latents@Wl[e]+bl[e].   d_out = [out (N x 256) | latents].
//
// v2: 32-token blocks (2 M-tiles) reuse each bf16 B-fragment twice from
// registers (halves L2 weight traffic); each wave owns complete OUTPUT tiles
// and computes the h / scale / shift accumulators itself, so the sine
// epilogue is fused in registers -- no f32 staging LDS, two barriers total.
// Accumulators start at 0 (inline-constant C) and biases fold into epilogue.
// ---------------------------------------------------------------------------

#define N_TOK   131072
#define IN_F    128
#define LAT_F   128
#define OUT_F   256
#define NEXP    8
#define OMEGA   30.0f
#define TOK_BLK 32            // tokens per block (2 WMMA M-tiles)

typedef __attribute__((ext_vector_type(16))) __bf16    v16bf;
typedef __attribute__((ext_vector_type(8)))  float     v8f;
typedef __attribute__((ext_vector_type(4)))  unsigned  uu4;

union Frag {                  // 8 dwords = 16 bf16 = one WMMA A/B operand
    unsigned u[8];
    uu4      q[2];
    v16bf    v;
};

// fp32 -> bf16 RNE, packed pair
__device__ __forceinline__ unsigned short f2bf(float x) {
    unsigned u = __builtin_bit_cast(unsigned, x);
    u += 0x7fffu + ((u >> 16) & 1u);
    return (unsigned short)(u >> 16);
}
__device__ __forceinline__ unsigned pack2(float a, float b) {
    return (unsigned)f2bf(a) | ((unsigned)f2bf(b) << 16);
}

// ---------------------------------------------------------------------------
// Kernel 1: convert + swizzle weights into B-fragment order.
// B fragment (16-bit, 32x16): lane l -> col n=l&15, K-half=(l>>4)*16,
// dword v holds K = khalf+2v, khalf+2v+1.  Frag base is 32B/lane contiguous.
// ---------------------------------------------------------------------------
__global__ void prep_weights(const float* __restrict__ W, const float* __restrict__ Wl,
                             unsigned* __restrict__ Wb, unsigned* __restrict__ Lb) {
    int t = blockIdx.x * blockDim.x + threadIdx.x;
    const int NW = NEXP * 16 * 4 * 32 * 8;   // 131072 dwords
    const int NL = NEXP * 32 * 4 * 32 * 8;   // 262144 dwords
    if (t < NW) {
        int v = t & 7, lane = (t >> 3) & 31, ks = (t >> 8) & 3, nt = (t >> 10) & 15, e = t >> 14;
        int n = nt * 16 + (lane & 15);
        int k = ks * 32 + ((lane >> 4) << 4) + v * 2;
        const float* w = W + (size_t)e * IN_F * OUT_F;
        Wb[t] = pack2(w[k * OUT_F + n], w[(k + 1) * OUT_F + n]);
    } else if (t < NW + NL) {
        int s = t - NW;
        int v = s & 7, lane = (s >> 3) & 31, ks = (s >> 8) & 3, nt = (s >> 10) & 31, e = s >> 15;
        int n = nt * 16 + (lane & 15);
        int k = ks * 32 + ((lane >> 4) << 4) + v * 2;
        const float* w = Wl + (size_t)e * LAT_F * (2 * OUT_F);
        Lb[s] = pack2(w[k * (2 * OUT_F) + n], w[(k + 1) * (2 * OUT_F) + n]);
    }
}

// ---------------------------------------------------------------------------
// Kernel 2: gating + top-2 softmax routing into per-expert lists.
// ---------------------------------------------------------------------------
__global__ void gate_route(const float* __restrict__ x, const float* __restrict__ gw,
                           const float* __restrict__ gb, int* __restrict__ cnt,
                           int* __restrict__ toks, float* __restrict__ wts) {
    int t = blockIdx.x * blockDim.x + threadIdx.x;
    if (t >= N_TOK) return;
    float acc[NEXP];
#pragma unroll
    for (int e = 0; e < NEXP; ++e) acc[e] = gb[e];
    const float* xr = x + (size_t)t * IN_F;
#pragma unroll 4
    for (int k = 0; k < IN_F; k += 4) {
        float4 xv = *(const float4*)(xr + k);
#pragma unroll
        for (int e = 0; e < NEXP; ++e)
            acc[e] += xv.x * gw[(k + 0) * NEXP + e] + xv.y * gw[(k + 1) * NEXP + e]
                    + xv.z * gw[(k + 2) * NEXP + e] + xv.w * gw[(k + 3) * NEXP + e];
    }
    int i0 = 0; float v0 = acc[0];
#pragma unroll
    for (int e = 1; e < NEXP; ++e) if (acc[e] > v0) { v0 = acc[e]; i0 = e; }
    int i1 = (i0 == 0) ? 1 : 0; float v1 = acc[i1];
#pragma unroll
    for (int e = 0; e < NEXP; ++e) if (e != i0 && acc[e] > v1) { v1 = acc[e]; i1 = e; }
    float ee = __expf(v1 - v0);
    float w0 = 1.0f / (1.0f + ee);
    float w1 = ee * w0;
    int p0 = atomicAdd(&cnt[i0], 1);
    toks[i0 * N_TOK + p0] = t;  wts[i0 * N_TOK + p0] = w0;
    int p1 = atomicAdd(&cnt[i1], 1);
    toks[i1 * N_TOK + p1] = t;  wts[i1 * N_TOK + p1] = w1;
}

// ---------------------------------------------------------------------------
// Kernel 3: per-expert WMMA GEMM with fused sine epilogue.
// Block = 256 thr = 8 wave32, 32 tokens.  Wave w owns out n-tiles {2w,2w+1};
// for each it computes h (W tile nt), scale (Wl tile nt), shift (Wl tile nt+16)
// over 2 M-tiles x 4 K-steps => 48 WMMAs/wave, epilogue fully in registers.
// ---------------------------------------------------------------------------
__global__ __launch_bounds__(256)
void expert_kernel(const float* __restrict__ x, const float* __restrict__ lat,
                   const float* __restrict__ b, const float* __restrict__ bl,
                   const unsigned* __restrict__ Wb, const unsigned* __restrict__ Lb,
                   const int* __restrict__ cnt, const int* __restrict__ toks,
                   const float* __restrict__ wts, float* __restrict__ out) {
    const int e     = blockIdx.y;
    const int tile  = blockIdx.x;
    const int count = cnt[e];
    if (tile * TOK_BLK >= count) return;      // block-uniform exit (EXEC stays full)

    __shared__ int      tok_s[TOK_BLK];
    __shared__ float    wt_s[TOK_BLK];
    __shared__ unsigned Xs[TOK_BLK * 64];     // 32 tokens x 128 K, packed bf16 (8 KB)
    __shared__ unsigned Ls[TOK_BLK * 64];     // (8 KB)

    const int tid = threadIdx.x;
    if (tid < TOK_BLK) {
        int slot = tile * TOK_BLK + tid;
        if (slot < count) { tok_s[tid] = toks[e * N_TOK + slot]; wt_s[tid] = wts[e * N_TOK + slot]; }
        else              { tok_s[tid] = toks[e * N_TOK + tile * TOK_BLK]; wt_s[tid] = 0.0f; }
    }
    __syncthreads();

    // Stage token tiles as packed bf16 (2048 dwords each, 8 per thread).
#pragma unroll
    for (int i = 0; i < 8; ++i) {
        int idx = tid + 256 * i;
        int m = idx >> 6, d = idx & 63;
        const float* xr = x   + (size_t)tok_s[m] * IN_F  + d * 2;
        const float* lr = lat + (size_t)tok_s[m] * LAT_F + d * 2;
        Xs[idx] = pack2(xr[0], xr[1]);
        Ls[idx] = pack2(lr[0], lr[1]);
    }
    __syncthreads();

    const int lane = tid & 31;
    const int l15  = lane & 15;               // A row within m-tile / C column
    const int asel = lane >> 4;
    const int wave = tid >> 5;

#pragma unroll
    for (int t2 = 0; t2 < 2; ++t2) {
        const int nt   = wave * 2 + t2;       // output n-tile 0..15
        const int jcol = nt * 16 + l15;       // output column

        // ---- h = X @ W[e] (both m-tiles share the B fragment) ----
        Frag faX[2][4];
#pragma unroll
        for (int mt = 0; mt < 2; ++mt)
#pragma unroll
            for (int ks = 0; ks < 4; ++ks)
#pragma unroll
                for (int v = 0; v < 8; ++v) {
                    int d = ks * 16 + asel * 8 + v;
                    faX[mt][ks].u[v] = Xs[(mt * 16 + l15) * 64 + d];
                }
        v8f accH[2] = {{0,0,0,0,0,0,0,0}, {0,0,0,0,0,0,0,0}};
#pragma unroll
        for (int ks = 0; ks < 4; ++ks) {
            Frag fb;
            const uu4* p = (const uu4*)(Wb + ((((size_t)e * 16 + nt) * 4 + ks) * 32 + lane) * 8);
            fb.q[0] = p[0]; fb.q[1] = p[1];
#pragma unroll
            for (int mt = 0; mt < 2; ++mt)
                accH[mt] = __builtin_amdgcn_wmma_f32_16x16x32_bf16(false, faX[mt][ks].v, false, fb.v,
                                                                   (short)0, accH[mt], false, false);
        }

        // ---- scale/shift = L @ Wl[e] tiles nt and nt+16 ----
        Frag faL[2][4];
#pragma unroll
        for (int mt = 0; mt < 2; ++mt)
#pragma unroll
            for (int ks = 0; ks < 4; ++ks)
#pragma unroll
                for (int v = 0; v < 8; ++v) {
                    int d = ks * 16 + asel * 8 + v;
                    faL[mt][ks].u[v] = Ls[(mt * 16 + l15) * 64 + d];
                }
        v8f accS[2] = {{0,0,0,0,0,0,0,0}, {0,0,0,0,0,0,0,0}};
        v8f accT[2] = {{0,0,0,0,0,0,0,0}, {0,0,0,0,0,0,0,0}};
#pragma unroll
        for (int ks = 0; ks < 4; ++ks) {
            Frag fbS, fbT;
            const uu4* ps = (const uu4*)(Lb + ((((size_t)e * 32 + nt)      * 4 + ks) * 32 + lane) * 8);
            const uu4* pt = (const uu4*)(Lb + ((((size_t)e * 32 + nt + 16) * 4 + ks) * 32 + lane) * 8);
            fbS.q[0] = ps[0]; fbS.q[1] = ps[1];
            fbT.q[0] = pt[0]; fbT.q[1] = pt[1];
#pragma unroll
            for (int mt = 0; mt < 2; ++mt) {
                accS[mt] = __builtin_amdgcn_wmma_f32_16x16x32_bf16(false, faL[mt][ks].v, false, fbS.v,
                                                                   (short)0, accS[mt], false, false);
                accT[mt] = __builtin_amdgcn_wmma_f32_16x16x32_bf16(false, faL[mt][ks].v, false, fbT.v,
                                                                   (short)0, accT[mt], false, false);
            }
        }

        // ---- fused epilogue: biases + sine + atomic combine ----
        const float bh = b [e * OUT_F     + jcol];
        const float bs = bl[e * 2 * OUT_F + jcol];
        const float bt = bl[e * 2 * OUT_F + OUT_F + jcol];
#pragma unroll
        for (int mt = 0; mt < 2; ++mt)
#pragma unroll
            for (int r = 0; r < 8; ++r) {
                int m = mt * 16 + r + asel * 8;              // C layout: M = r + 8*(lane>>4)
                float hv = accH[mt][r] + bh;
                float sc = accS[mt][r] + bs;
                float sh = accT[mt][r] + bt;
                float val = wt_s[m] * __sinf(OMEGA * hv * sc + sh);
                atomicAdd(out + (size_t)tok_s[m] * OUT_F + jcol, val);
            }
    }
}

// ---------------------------------------------------------------------------
extern "C" void kernel_launch(void* const* d_in, const int* in_sizes, int n_in,
                              void* d_out, int out_size, void* d_ws, size_t ws_size,
                              hipStream_t stream) {
    (void)in_sizes; (void)n_in; (void)out_size; (void)ws_size;
    const float* x   = (const float*)d_in[0];
    const float* lat = (const float*)d_in[1];
    const float* gw  = (const float*)d_in[2];
    const float* gb  = (const float*)d_in[3];
    const float* W   = (const float*)d_in[4];
    const float* b   = (const float*)d_in[5];
    const float* Wl  = (const float*)d_in[6];
    const float* bl  = (const float*)d_in[7];
    float* out = (float*)d_out;

    // Workspace (~9.9 MB): counters | token lists | weight lists | bf16 W | bf16 Wl
    char* ws = (char*)d_ws;
    int*      cnt  = (int*)ws;
    int*      toks = (int*)(ws + 256);
    float*    wts  = (float*)(ws + 256 + (size_t)NEXP * N_TOK * 4);
    unsigned* Wb   = (unsigned*)(ws + 256 + (size_t)NEXP * N_TOK * 8);
    unsigned* Lb   = Wb + NEXP * 16 * 4 * 32 * 8;

    hipMemsetAsync(cnt, 0, 256, stream);
    hipMemsetAsync(out, 0, (size_t)N_TOK * OUT_F * sizeof(float), stream);
    hipMemcpyAsync(out + (size_t)N_TOK * OUT_F, lat,
                   (size_t)N_TOK * LAT_F * sizeof(float),
                   hipMemcpyDeviceToDevice, stream);

    prep_weights<<<(NEXP * 48 * 4 * 32 * 8 + 255) / 256, 256, 0, stream>>>(W, Wl, Wb, Lb);
    gate_route<<<N_TOK / 256, 256, 0, stream>>>(x, gw, gb, cnt, toks, wts);
    dim3 grid(N_TOK / TOK_BLK, NEXP);
    expert_kernel<<<grid, 256, 0, stream>>>(x, lat, b, bl, Wb, Lb, cnt, toks, wts, out);
}
